// BGFG_XYZT_70153995813579
// MI455X (gfx1250) — matrix operand order repaired and checked by reading, not compiled, not run on hardware
//
#include <hip/hip_runtime.h>
#include <cmath>
#include <cstdint>

typedef __attribute__((ext_vector_type(16))) _Float16 v16h;
typedef __attribute__((ext_vector_type(8)))  _Float16 v8h;
typedef __attribute__((ext_vector_type(4)))  _Float16 v4h;
typedef __attribute__((ext_vector_type(2)))  _Float16 v2h;
typedef __attribute__((ext_vector_type(8)))  float    v8f;
typedef _Float16 h16;

#define WAVES        4
#define THREADS      (WAVES * 32)
#define PTS_PER_BLK  (WAVES * 16)

// Packed weight image (halves). Fragment f = kchunk*ntiles + ntile, 512 halves
// each, laid out frag[lane*16 + j] so a lane reads 32 B contiguous (2x b128).
#define OFF_S0 0       // 1 kchunk x 4 ntiles
#define OFF_S1 2048    // 2 x 4
#define OFF_S2 6144    // 2 x 2 (17 cols padded to 32)
#define OFF_C0 8192    // 1 x 4 (18/15 rows padded to 32)
#define OFF_C1 10240   // 2 x 4
#define OFF_C2 14336   // 2 x 4
#define OFF_C3 18432   // 2 x 1 (3 cols padded to 16)
#define W_TOTAL 19456  // halves per branch image (38912 B)
#define W_BYTES (W_TOTAL * 2)
#define FRAG_H 512

#define SSTR 72   // f16 activation staging stride (halves); 144 B, 16 B aligned
#define HSTR 36   // f32 head staging stride (floats)

struct Params {
  const float* emb[3];
  const float* views;
  const float* mask[3];
  const h16*   wpacked;   // 3 * W_TOTAL halves in d_ws
  float* out;
};

__device__ __forceinline__ v8f wmma_f16(v16h a, v16h b, v8f c) {
  return __builtin_amdgcn_wmma_f32_16x16x32_f16(
      false, a, false, b, (short)0, c, false, false);
}

__device__ __forceinline__ v16h cat8(v8h lo, v8h hi) {
  return __builtin_shufflevector(lo, hi, 0, 1, 2, 3, 4, 5, 6, 7,
                                         8, 9, 10, 11, 12, 13, 14, 15);
}

// Packed B fragment: 32 B contiguous per lane -> 2x ds_load_b128
__device__ __forceinline__ v16h load_b(const h16* frag, int lane) {
  const v8h* pp = (const v8h*)(frag + lane * 16);
  return cat8(pp[0], pp[1]);
}

// A fragment (16x32 f16, ISA 7.12.2): lane holds row m = lane&15; halves are
// two contiguous 8-half runs (K = base+0..7, base+16..23), both 16 B aligned.
__device__ __forceinline__ v16h load_a(const h16* s, int kc, int lane) {
  const int row  = lane & 15;
  const int kadd = (lane >= 16) ? 8 : 0;
  const h16* base = s + row * SSTR + kc + kadd;
  return cat8(*(const v8h*)base, *(const v8h*)(base + 16));
}

// D (16x16 f32): lane n = lane&15; VGPR r -> row m = r + (lane>=16 ? 8 : 0).
// Convert to f16 first (v_cvt_pk_f16_f32 pairs), relu as packed half max
// (v_pk_max_num_f16) -- max(.,0) commutes with the monotonic f32->f16 cvt.
__device__ __forceinline__ void store_relu(h16* s, v8f d, int ntile, int lane) {
  const int n     = ntile * 16 + (lane & 15);
  const int mbase = (lane >= 16) ? 8 : 0;
#pragma unroll
  for (int r = 0; r < 8; r += 2) {
    v2h hv = { (h16)d[r], (h16)d[r + 1] };
    v2h z  = { (h16)0.f, (h16)0.f };
    hv = __builtin_elementwise_max(hv, z);
    s[(mbase + r)     * SSTR + n] = hv[0];
    s[(mbase + r + 1) * SSTR + n] = hv[1];
  }
}

// ---------------- main kernel ------------------------------------------------
__global__ __launch_bounds__(THREADS) void nerf3_wmma_kernel(Params p) {
  __shared__ alignas(16) h16   sW[W_TOTAL];
  __shared__ alignas(16) h16   sAct[WAVES][16 * SSTR];
  __shared__ float sHead[WAVES][16 * HSTR];
  __shared__ float sSig[PTS_PER_BLK];
  __shared__ float sCol[PTS_PER_BLK][3];

  const int tid      = threadIdx.x;
  const int wave     = tid >> 5;
  const int lane     = tid & 31;
  const int blkBase  = blockIdx.x * PTS_PER_BLK;
  const int waveBase = blkBase + wave * 16;

  for (int i = tid; i < PTS_PER_BLK; i += THREADS) {
    sSig[i] = 0.f; sCol[i][0] = 0.f; sCol[i][1] = 0.f; sCol[i][2] = 0.f;
  }

  h16*   act  = sAct[wave];
  float* head = sHead[wave];
  const unsigned ldsW = (unsigned)(uintptr_t)(void*)&sW[0]; // LDS byte offset

  for (int b = 0; b < 3; ++b) {
    __syncthreads();  // previous branch done reading sW (b==0: accum init)

    // ---- async DMA: packed branch weights (38912 B) global -> LDS ----
    {
      const h16* wsrc = p.wpacked + (size_t)b * W_TOTAL;
#pragma unroll
      for (int c = 0; c < 19; ++c) {
        const unsigned byteOff = (unsigned)(c * THREADS + tid) * 16u;
        const unsigned lds = ldsW + byteOff;
        asm volatile("global_load_async_to_lds_b128 %0, %1, %2"
                     :: "v"(lds), "v"(byteOff), "s"(wsrc)
                     : "memory");
      }
      asm volatile("s_wait_asynccnt 0x0" ::: "memory");
    }

    // ---- coalesced embedding load: 16 pts x 32 f32 -> f16 staging ----
    {
      const float4* embv = (const float4*)(p.emb[b] + (size_t)waveBase * 32);
#pragma unroll
      for (int t = 0; t < 4; ++t) {
        int fi = lane + t * 32;
        float4 v = embv[fi];
        int row = fi >> 3;
        int col = (fi & 7) * 4;
        v4h hv = { (h16)v.x, (h16)v.y, (h16)v.z, (h16)v.w };
        *(v4h*)(act + row * SSTR + col) = hv;   // ds_store_b64
      }
    }
    __syncthreads();  // weights visible to all waves

    // ---- sigma L0: (16x32) @ (32x64), relu ----
    {
      v16h a = load_a(act, 0, lane);
#pragma unroll
      for (int nt = 0; nt < 4; ++nt) {
        v8f c = {};
        c = wmma_f16(a, load_b(sW + OFF_S0 + nt * FRAG_H, lane), c);
        store_relu(act, c, nt, lane);
      }
    }
    // ---- sigma L1: (16x64) @ (64x64), relu ----
    {
      v16h a0 = load_a(act, 0, lane);
      v16h a1 = load_a(act, 32, lane);
#pragma unroll
      for (int nt = 0; nt < 4; ++nt) {
        v8f c = {};
        c = wmma_f16(a0, load_b(sW + OFF_S1 + nt * FRAG_H,       lane), c);
        c = wmma_f16(a1, load_b(sW + OFF_S1 + (4 + nt) * FRAG_H, lane), c);
        store_relu(act, c, nt, lane);
      }
    }
    // ---- sigma head: (16x64) @ (64x32pad), raw -> f32 head staging ----
    {
      v16h a0 = load_a(act, 0, lane);
      v16h a1 = load_a(act, 32, lane);
      const int nloc  = lane & 15;
      const int mbase = (lane >= 16) ? 8 : 0;
#pragma unroll
      for (int nt = 0; nt < 2; ++nt) {
        v8f c = {};
        c = wmma_f16(a0, load_b(sW + OFF_S2 + nt * FRAG_H,       lane), c);
        c = wmma_f16(a1, load_b(sW + OFF_S2 + (2 + nt) * FRAG_H, lane), c);
        const int n = nt * 16 + nloc;
#pragma unroll
        for (int r = 0; r < 8; ++r) head[(mbase + r) * HSTR + n] = c[r];
      }
    }
    // ---- color L0: build ci = [views, geo] (bg) or geo, K padded to 32 ----
    {
      v16h a;
      const int row  = lane & 15;
      const int kadd = (lane >= 16) ? 8 : 0;
      const float* vw = p.views + (size_t)(waveBase + row) * 3;
#pragma unroll
      for (int j = 0; j < 16; ++j) {
        int k = j + ((j >= 8) ? 8 : 0) + kadd;
        float v = 0.f;  // explicit zero for padded K (avoid 0*Inf NaN in WMMA)
        if (b == 0) {
          if (k < 3)       v = vw[k];
          else if (k < 18) v = head[row * HSTR + (k - 1)];
        } else {
          if (k < 15)      v = head[row * HSTR + (2 + k)];
        }
        a[j] = (h16)v;
      }
#pragma unroll
      for (int nt = 0; nt < 4; ++nt) {
        v8f c = {};
        c = wmma_f16(a, load_b(sW + OFF_C0 + nt * FRAG_H, lane), c);
        store_relu(act, c, nt, lane);
      }
    }
    // ---- color L1 / L2: (16x64) @ (64x64), relu ----
    for (int l = 0; l < 2; ++l) {
      const int off = (l == 0) ? OFF_C1 : OFF_C2;
      v16h a0 = load_a(act, 0, lane);
      v16h a1 = load_a(act, 32, lane);
#pragma unroll
      for (int nt = 0; nt < 4; ++nt) {
        v8f c = {};
        c = wmma_f16(a0, load_b(sW + off + nt * FRAG_H,       lane), c);
        c = wmma_f16(a1, load_b(sW + off + (4 + nt) * FRAG_H, lane), c);
        store_relu(act, c, nt, lane);
      }
    }
    // ---- color head: (16x64) @ (64x16pad), raw cols 0..2 -> head[17..19] ----
    {
      v16h a0 = load_a(act, 0, lane);
      v16h a1 = load_a(act, 32, lane);
      v8f c = {};
      c = wmma_f16(a0, load_b(sW + OFF_C3,          lane), c);
      c = wmma_f16(a1, load_b(sW + OFF_C3 + FRAG_H, lane), c);
      const int n     = lane & 15;
      const int mbase = (lane >= 16) ? 8 : 0;
      if (n < 3) {
#pragma unroll
        for (int r = 0; r < 8; ++r) head[(mbase + r) * HSTR + 17 + n] = c[r];
      }
    }
    // ---- per-point finalize + accumulate (lanes 0..15, one point each) ----
    if (lane < 16) {
      const int q  = lane;
      const int pg = waveBase + q;
      const float m  = p.mask[b][pg];
      const float x  = head[q * HSTR + 0];
      const float sp = (x > 20.f) ? x : log1pf(__expf(x));
      const float s  = sp * m;
      const int qb = wave * 16 + q;
      sSig[qb] += s;
#pragma unroll
      for (int c3 = 0; c3 < 3; ++c3) {
        float cr = head[q * HSTR + 17 + c3];
        float sg = 1.f / (1.f + __expf(-cr));
        sCol[qb][c3] += s * (sg * m);
      }
    }
  }  // branch loop

  __syncthreads();
  // ---- output: [color0..2, sigma] per point ----
  for (int i = tid; i < PTS_PER_BLK * 4; i += THREADS) {
    const int q = i >> 2, c = i & 3;
    const float st = sSig[q] + 1e-9f;
    const float v  = (c < 3) ? (sCol[q][c] / st) : st;
    p.out[(size_t)(blkBase + q) * 4 + c] = v;
  }
}

extern "C" void kernel_launch(void* const* d_in, const int* in_sizes, int n_in,
                              void* d_out, int out_size, void* d_ws, size_t ws_size,
                              hipStream_t stream) {
  (void)n_in; (void)out_size; (void)ws_size;

  // d_ws layout: [0, 3*W_BYTES) packed f16 weights
  h16* wpacked = (h16*)d_ws;

  struct WTab { const float* w[21]; };
  static_assert(sizeof(WTab) == 21 * sizeof(void*), "");

  // One-time (per launch) weight pack: f32 -> packed f16 B-fragments.
  struct Launcher {
    static __global__ __launch_bounds__(256) void pack(WTab t, h16* dst) {
      const int b = blockIdx.x;                 // branch 0..2
      const int cin = (b == 0) ? 18 : 15;
      const int rows[7]  = {32, 64, 64, cin, 64, 64, 64};
      const int cols[7]  = {64, 64, 17, 64, 64, 64, 3};
      const int nns[7]   = {4, 4, 2, 4, 4, 4, 1};
      const int nfr[7]   = {4, 8, 4, 4, 8, 8, 2};
      const int offs[7]  = {OFF_S0, OFF_S1, OFF_S2, OFF_C0, OFF_C1, OFF_C2, OFF_C3};
      h16* out = dst + (size_t)b * W_TOTAL;
#pragma unroll
      for (int m = 0; m < 7; ++m) {
        const float* src = t.w[b * 7 + m];
        const int total = nfr[m] * FRAG_H;
        for (int i = threadIdx.x; i < total; i += 256) {
          const int f   = i >> 9;
          const int rem = i & 511;
          const int ln  = rem >> 4;             // fragment lane slot 0..31
          const int j   = rem & 15;             // half slot 0..15
          const int kci = f / nns[m];
          const int nt  = f % nns[m];
          // B layout: lanes 0-15 hold K=kc+0..7,16..23; lanes 16-31 K=kc+8..15,24..31
          const int k = kci * 32 + j + ((j & 8) ? 8 : 0) + ((ln & 16) ? 8 : 0);
          const int n = nt * 16 + (ln & 15);
          float v = (k < rows[m] && n < cols[m]) ? src[k * cols[m] + n] : 0.f;
          out[offs[m] + i] = (h16)v;
        }
      }
    }
  };

  WTab t;
  for (int i = 0; i < 21; ++i) t.w[i] = (const float*)d_in[7 + i];
  hipLaunchKernelGGL(Launcher::pack, dim3(3), dim3(256), 0, stream, t, wpacked);

  Params p;
  p.emb[0] = (const float*)d_in[0];
  p.emb[1] = (const float*)d_in[1];
  p.emb[2] = (const float*)d_in[2];
  p.views  = (const float*)d_in[3];
  p.mask[0] = (const float*)d_in[4];
  p.mask[1] = (const float*)d_in[5];
  p.mask[2] = (const float*)d_in[6];
  p.wpacked = wpacked;
  p.out = (float*)d_out;

  const int N    = in_sizes[4];
  const int grid = N / PTS_PER_BLK;
  hipLaunchKernelGGL(nerf3_wmma_kernel, dim3(grid), dim3(THREADS), 0, stream, p);
}
